// PowerGConvDyn_12524124635992
// MI455X (gfx1250) — compile-verified, matrix operand order checked by reading, not run on hardware
//
#include <hip/hip_runtime.h>

typedef __attribute__((ext_vector_type(16))) __bf16 v16bf;
typedef __attribute__((ext_vector_type(8)))  float  v8f;

#define D 128  // feature width (fixed by the reference)

// ---------------------------------------------------------------------------
// Zero-fill (float4 vectorized) — Z accumulators must start at 0 each call.
// ---------------------------------------------------------------------------
__global__ __launch_bounds__(256) void zero_f4(float* __restrict__ p, long n4) {
  long i = (long)blockIdx.x * blockDim.x + threadIdx.x;
  if (i < n4) {
    float4 z; z.x = 0.f; z.y = 0.f; z.z = 0.f; z.w = 0.f;
    ((float4*)p)[i] = z;
  }
}

// ---------------------------------------------------------------------------
// COO SpMM: Zout[row[e]] += val[e] * Zin[col[e]]
// One wave32 per edge; lane owns a float4 chunk of the 128-wide feature row.
// Zin/Zout are 51.2 MB each -> L2 (192 MB) resident; atomics resolve in L2.
// ---------------------------------------------------------------------------
__global__ __launch_bounds__(256) void spmm_coo(
    const int*   __restrict__ row,
    const int*   __restrict__ col,
    const float* __restrict__ val,
    const float* __restrict__ Zin,
    float*       __restrict__ Zout,
    int E)
{
  int e = blockIdx.x * 8 + (threadIdx.x >> 5);
  if (e >= E) return;
  int lane = threadIdx.x & 31;

  int   r = row[e];
  int   c = col[e];
  float v = val[e];

  float4 z = *(const float4*)(Zin + (size_t)c * D + lane * 4);
  float* dst = Zout + (size_t)r * D + lane * 4;

  unsafeAtomicAdd(dst + 0, v * z.x);
  unsafeAtomicAdd(dst + 1, v * z.y);
  unsafeAtomicAdd(dst + 2, v * z.z);
  unsafeAtomicAdd(dst + 3, v * z.w);
}

// ---------------------------------------------------------------------------
// out[N,128] (+)= Z[N,128] @ W[wRowOff:wRowOff+128, 0:128]   (bf16 WMMA, f32 acc)
// Block = 256 threads = 8 waves; wave w owns output column tile w (16 cols).
// Each wave preloads its 128x16 B panel as 4 v16bf fragments (register
// resident, W is L2-hot), then streams tilesPerBlock row tiles.
// initWithBias=1: C seeded with bias (first hop); else C = current out (RMW).
// ---------------------------------------------------------------------------
__global__ __launch_bounds__(256) void gemm_accum_wmma(
    const float* __restrict__ Z,
    const float* __restrict__ W,      // [512,128] row-major, rows = K index
    int wRowOff,
    const float* __restrict__ bias,   // [128]
    float*       __restrict__ out,    // [N,128]
    int rowTiles, int tilesPerBlock, int initWithBias)
{
  const int lane    = threadIdx.x & 31;
  const int wave    = threadIdx.x >> 5;   // 0..7 -> column tile
  const int half    = lane >> 4;          // 0 or 1
  const int l16     = lane & 15;
  const int col     = wave * 16 + l16;

  // ---- Preload B fragments: B[k, col], 32-deep K chunks, WMMA B layout ----
  v16bf bfrag[4];
  #pragma unroll
  for (int kc4 = 0; kc4 < 4; ++kc4) {
    const int kc = kc4 * 32;
    #pragma unroll
    for (int i = 0; i < 8; ++i) {
      bfrag[kc4][i]     = (__bf16)W[(size_t)(wRowOff + kc +      half * 8 + i) * D + col];
      bfrag[kc4][8 + i] = (__bf16)W[(size_t)(wRowOff + kc + 16 + half * 8 + i) * D + col];
    }
  }

  for (int t = 0; t < tilesPerBlock; ++t) {
    int rowTile = blockIdx.x * tilesPerBlock + t;
    if (rowTile >= rowTiles) return;
    int rowBase = rowTile * 16;
    const float* zrow = Z + (size_t)(rowBase + l16) * D;

    // ---- C fragment: seed with bias or accumulate into existing out ----
    v8f c;
    if (initWithBias) {
      float bb = bias[col];
      #pragma unroll
      for (int r = 0; r < 8; ++r) c[r] = bb;
    } else {
      #pragma unroll
      for (int r = 0; r < 8; ++r)
        c[r] = out[(size_t)(rowBase + half * 8 + r) * D + col];
    }

    // ---- 4 x v_wmma_f32_16x16x32_bf16 over K=128 ----
    #pragma unroll
    for (int kc4 = 0; kc4 < 4; ++kc4) {
      const int kc = kc4 * 32;
      const float* p0 = zrow + kc +      half * 8;
      const float* p1 = zrow + kc + 16 + half * 8;
      float4 x0 = *(const float4*)(p0);
      float4 x1 = *(const float4*)(p0 + 4);
      float4 x2 = *(const float4*)(p1);
      float4 x3 = *(const float4*)(p1 + 4);

      v16bf a;
      a[0]  = (__bf16)x0.x; a[1]  = (__bf16)x0.y; a[2]  = (__bf16)x0.z; a[3]  = (__bf16)x0.w;
      a[4]  = (__bf16)x1.x; a[5]  = (__bf16)x1.y; a[6]  = (__bf16)x1.z; a[7]  = (__bf16)x1.w;
      a[8]  = (__bf16)x2.x; a[9]  = (__bf16)x2.y; a[10] = (__bf16)x2.z; a[11] = (__bf16)x2.w;
      a[12] = (__bf16)x3.x; a[13] = (__bf16)x3.y; a[14] = (__bf16)x3.z; a[15] = (__bf16)x3.w;

      c = __builtin_amdgcn_wmma_f32_16x16x32_bf16(
              /*neg_a=*/false, a, /*neg_b=*/false, bfrag[kc4],
              /*c_mod=*/(short)0, c, /*reuse_a=*/false, /*reuse_b=*/false);
    }

    // ---- Store D fragment ----
    #pragma unroll
    for (int r = 0; r < 8; ++r)
      out[(size_t)(rowBase + half * 8 + r) * D + col] = c[r];
  }
}

// ---------------------------------------------------------------------------
// Driver: out = [X | A@X | A^2@X | A^3@X] @ W + b, fused as per-hop accumulate
// ---------------------------------------------------------------------------
extern "C" void kernel_launch(void* const* d_in, const int* in_sizes, int n_in,
                              void* d_out, int out_size, void* d_ws, size_t ws_size,
                              hipStream_t stream) {
  const float* X  = (const float*)d_in[0];
  const int*   Ar = (const int*)  d_in[1];
  const int*   Ac = (const int*)  d_in[2];
  const float* Av = (const float*)d_in[3];
  const float* W  = (const float*)d_in[4];
  const float* b  = (const float*)d_in[5];
  float* out = (float*)d_out;

  const int N = in_sizes[0] / D;     // 100000
  const int E = in_sizes[1];         // 3200000

  float* Z1 = (float*)d_ws;                  // ping
  float* Z2 = Z1 + (size_t)N * D;            // pong (2 x 51.2 MB in d_ws)

  const int rowTiles = (N + 15) / 16;        // 6250 (exact: N % 16 == 0)
  const int TPB = 8;                         // row tiles per block
  const int gemmBlocks = (rowTiles + TPB - 1) / TPB;
  const long n4 = (long)N * D / 4;
  const int zeroBlocks = (int)((n4 + 255) / 256);
  const int spmmBlocks = (E + 7) / 8;

  // hop 0: out = X @ W[0:128] + b
  gemm_accum_wmma<<<gemmBlocks, 256, 0, stream>>>(X,  W,   0, b, out, rowTiles, TPB, 1);

  // hop 1: Z1 = A @ X ; out += Z1 @ W[128:256]
  zero_f4<<<zeroBlocks, 256, 0, stream>>>(Z1, n4);
  spmm_coo<<<spmmBlocks, 256, 0, stream>>>(Ar, Ac, Av, X,  Z1, E);
  gemm_accum_wmma<<<gemmBlocks, 256, 0, stream>>>(Z1, W, 128, b, out, rowTiles, TPB, 0);

  // hop 2: Z2 = A @ Z1 ; out += Z2 @ W[256:384]
  zero_f4<<<zeroBlocks, 256, 0, stream>>>(Z2, n4);
  spmm_coo<<<spmmBlocks, 256, 0, stream>>>(Ar, Ac, Av, Z1, Z2, E);
  gemm_accum_wmma<<<gemmBlocks, 256, 0, stream>>>(Z2, W, 256, b, out, rowTiles, TPB, 0);

  // hop 3: Z1 = A @ Z2 ; out += Z1 @ W[384:512]
  zero_f4<<<zeroBlocks, 256, 0, stream>>>(Z1, n4);
  spmm_coo<<<spmmBlocks, 256, 0, stream>>>(Ar, Ac, Av, Z2, Z1, E);
  gemm_accum_wmma<<<gemmBlocks, 256, 0, stream>>>(Z1, W, 384, b, out, rowTiles, TPB, 0);
}
